// PointerNetwork_22728966930984
// MI455X (gfx1250) — matrix operand order repaired and checked by reading, not compiled, not run on hardware
//
#include <hip/hip_runtime.h>
#include <hip/hip_bf16.h>
#include <math.h>

// Problem constants (reference: B=16, T=256, D=128, H=256)
#define BB 16
#define TT 256
#define DD 128
#define HH 256
#define H3 768   // 3*H
#define TQ 8     // query rows per attention block

typedef __attribute__((ext_vector_type(16))) __bf16 v16bf;
typedef __attribute__((ext_vector_type(8)))  __bf16 v8bf;
typedef __attribute__((ext_vector_type(8)))  float  v8f;
typedef __attribute__((ext_vector_type(4)))  float  v4f;

// Native v_tanh_f32 on gfx1250 when the builtin exists; OCML fallback.
__device__ __forceinline__ float fast_tanh(float x) {
#if __has_builtin(__builtin_amdgcn_tanhf)
    return __builtin_amdgcn_tanhf(x);
#elif __has_builtin(__builtin_amdgcn_tanh_f32)
    return __builtin_amdgcn_tanh_f32(x);
#else
    return tanhf(x);
#endif
}

__device__ __forceinline__ float sigmoidf_(float x) {
    return 0.5f * (1.0f + fast_tanh(0.5f * x));
}

// ---------------------------------------------------------------------------
// GEMM: C[M,N] = A[M,K](f32) @ W[N,K]^T(f32) + bias[N], bf16 WMMA, f32 acc.
// One wave computes a 16x64 C strip (4 N-tiles) sharing one A operand load.
// Per k-step, ALL raw f32 data (4 A-vec4 + 16 B-vec4, independent b128 loads)
// is staged into distinct registers before any conversion, so the compiler
// can batch the loads and drain them with descending s_wait_loadcnt instead
// of serializing; the 4 WMMAs then issue back-to-back on independent accs.
// WMMA wave32 layouts (CDNA5 ISA 7.12.2):
//   A 16x32 bf16: lane m=(l&15); lanes<16: elems 0..7 -> K 0..7, 8..15 -> K16..23
//                 lanes>=16: elems 0..7 -> K 8..15, 8..15 -> K 24..31
//   B 32x16 bf16: lane n=(l&15); elems 0..15 hold the same K as A's elems 0..15
//   C 16x16 f32:  lane n=(l&15); vgpr v -> row v (+8 for upper lane half)
// ---------------------------------------------------------------------------
__global__ __launch_bounds__(256) void gemm_bf16_wmma(
    const float* __restrict__ A, const float* __restrict__ W,
    const float* __restrict__ bias, float* __restrict__ C,
    int M, int N, int K, const int* __restrict__ gather, int transposedOut)
{
    const int lane  = threadIdx.x & 31;
    const int wave  = threadIdx.x >> 5;
    const int tilesN4 = N >> 6;                 // groups of 4 N-tiles
    const int group = blockIdx.x * 8 + wave;
    const int tm  = group / tilesN4;
    const int tng = group - tm * tilesN4;

    int arow = tm * 16 + (lane & 15);
    if (gather) arow = gather[arow];
    const float* __restrict__ Arow = A + (size_t)arow * (size_t)K;

    const float* __restrict__ Wrow[4];
    #pragma unroll
    for (int c = 0; c < 4; ++c)
        Wrow[c] = W + (size_t)(tng * 64 + c * 16 + (lane & 15)) * (size_t)K;

    const int khalfA = (lane & 16) ? 8 : 0;
    const int khalfB = (lane & 16) ? 8 : 0;

    v8f acc[4] = {{}, {}, {}, {}};
    const int KK = K >> 5;
    for (int kk = 0; kk < KK; ++kk) {
        const int abase = kk * 32 + khalfA;
        const int bbase = kk * 32 + khalfB;

        // ---- stage: issue all independent 128-bit loads first ----
        v4f ar[4];
        ar[0] = *(const v4f*)(Arow + abase);
        ar[1] = *(const v4f*)(Arow + abase + 4);
        ar[2] = *(const v4f*)(Arow + abase + 16);
        ar[3] = *(const v4f*)(Arow + abase + 20);
        v4f br[4][4];
        #pragma unroll
        for (int c = 0; c < 4; ++c) {
            #pragma unroll
            for (int w = 0; w < 4; ++w)
                br[c][w] = *(const v4f*)(Wrow[c] + bbase + (w & 1) * 4 + (w >> 1) * 16);
        }

        // ---- convert ----
        v16bf a;
        #pragma unroll
        for (int w = 0; w < 4; ++w) {
            #pragma unroll
            for (int e = 0; e < 4; ++e)
                a[w * 4 + e] = (__bf16)ar[w][e];
        }
        v16bf b[4];
        #pragma unroll
        for (int c = 0; c < 4; ++c) {
            #pragma unroll
            for (int w = 0; w < 4; ++w) {
                #pragma unroll
                for (int e = 0; e < 4; ++e)
                    b[c][w * 4 + e] = (__bf16)br[c][w][e];
            }
        }

        // ---- 4 back-to-back WMMAs, shared A ----
        #pragma unroll
        for (int c = 0; c < 4; ++c)
            acc[c] = __builtin_amdgcn_wmma_f32_16x16x32_bf16(
                false, a, false, b[c], (short)0, acc[c], false, false);
    }

    const int mbase = tm * 16 + ((lane & 16) ? 8 : 0);
    #pragma unroll
    for (int c = 0; c < 4; ++c) {
        const int nn = tng * 64 + c * 16 + (lane & 15);
        const float bv = bias ? bias[nn] : 0.0f;
        #pragma unroll
        for (int v = 0; v < 8; ++v) {
            const int m = mbase + v;
            const float val = acc[c][v] + bv;
            if (transposedOut) {
                // kT[b][a][s]: b = m/256, a = nn, s = m%256  (requires N==256)
                C[(size_t)(m >> 8) * (size_t)(HH * TT) + (size_t)nn * TT + (m & 255)] = val;
            } else {
                C[(size_t)m * (size_t)N + nn] = val;
            }
        }
    }
}

// ---------------------------------------------------------------------------
// Persistent GRU scan. One workgroup, 24 waves (768 threads).
// Whh (768x256) held in registers as bf16 WMMA B-operands: each wave owns
// 2 N-tiles x 8 K-slices = 16 x v16bf = 128 VGPRs, resident for all T steps.
// h lives in LDS (f32 + bf16 mirror); gh strip + bhh cached in LDS.
// Per step: 16 WMMAs/wave -> gh, barrier, fused GRU gates, barrier.
// ---------------------------------------------------------------------------
__global__ __launch_bounds__(768, 1) void gru_scan(
    const float* __restrict__ gi,   // [B*T, 3H] precomputed x@Wih^T + bih
    const float* __restrict__ Whh,  // [3H, H]
    const float* __restrict__ bhh,  // [3H]
    const float* __restrict__ h0,   // [B, H] or nullptr -> zeros
    float* __restrict__ out,        // [B*T, H]
    float* __restrict__ hT)         // [B, H]
{
    extern __shared__ char smem[];
    float*  hf  = (float*)smem;                // 16*256 f32   (16 KB)
    float*  gh  = hf + BB * HH;                // 16*768 f32   (48 KB)
    float*  bhs = gh + BB * H3;                // 768 f32      (3 KB)
    __bf16* hb  = (__bf16*)(bhs + H3);         // 16*256 bf16  (8 KB)

    const int tid  = threadIdx.x;
    const int lane = tid & 31;
    const int wave = tid >> 5;                 // 0..23 -> N-tiles 2w, 2w+1

    // Preload Whh B-operands into registers (f32 -> bf16 once).
    const int khalfB = (lane & 16) ? 16 : 0;
    v16bf Breg[2][8];
    #pragma unroll
    for (int tI = 0; tI < 2; ++tI) {
        const int n = (wave * 2 + tI) * 16 + (lane & 15);
        const float* __restrict__ Wr = Whh + (size_t)n * HH;
        #pragma unroll
        for (int kk = 0; kk < 8; ++kk) {
            v4f w0 = *(const v4f*)(Wr + kk * 32 + khalfB);
            v4f w1 = *(const v4f*)(Wr + kk * 32 + khalfB + 4);
            v4f w2 = *(const v4f*)(Wr + kk * 32 + khalfB + 8);
            v4f w3 = *(const v4f*)(Wr + kk * 32 + khalfB + 12);
            #pragma unroll
            for (int e = 0; e < 4; ++e) {
                Breg[tI][kk][e]      = (__bf16)w0[e];
                Breg[tI][kk][4 + e]  = (__bf16)w1[e];
                Breg[tI][kk][8 + e]  = (__bf16)w2[e];
                Breg[tI][kk][12 + e] = (__bf16)w3[e];
            }
        }
    }

    // Init hidden state + cache bhh.
    for (int idx = tid; idx < BB * HH; idx += 768) {
        const float v0 = h0 ? h0[idx] : 0.0f;
        hf[idx] = v0;
        hb[idx] = (__bf16)v0;
    }
    if (tid < H3) bhs[tid] = bhh[tid];
    __syncthreads();

    const int mrowA  = lane & 15;
    const int khalfA = (lane & 16) ? 16 : 0;
    const int mb     = (lane & 16) ? 8 : 0;
    const int nc0    = (wave * 2 + 0) * 16 + (lane & 15);
    const int nc1    = (wave * 2 + 1) * 16 + (lane & 15);

    for (int t = 0; t < TT; ++t) {
        // gh = h @ Whh^T  (M=16 batch, N=768, K=256)
        v8f acc0 = {}, acc1 = {};
        #pragma unroll
        for (int kk = 0; kk < 8; ++kk) {
            const __bf16* __restrict__ hp = hb + mrowA * HH + kk * 32 + khalfA;
            const v8bf lo = *(const v8bf*)(hp);
            const v8bf hi = *(const v8bf*)(hp + 8);
            const v16bf a = __builtin_shufflevector(
                lo, hi, 0,1,2,3,4,5,6,7,8,9,10,11,12,13,14,15);
            acc0 = __builtin_amdgcn_wmma_f32_16x16x32_bf16(
                false, a, false, Breg[0][kk], (short)0, acc0, false, false);
            acc1 = __builtin_amdgcn_wmma_f32_16x16x32_bf16(
                false, a, false, Breg[1][kk], (short)0, acc1, false, false);
        }
        #pragma unroll
        for (int v = 0; v < 8; ++v) {
            gh[(mb + v) * H3 + nc0] = acc0[v];
            gh[(mb + v) * H3 + nc1] = acc1[v];
        }
        __syncthreads();

        // Fused GRU gates + h update + output store.
        for (int idx = tid; idx < BB * HH; idx += 768) {
            const int b = idx >> 8;
            const int j = idx & 255;
            const float* __restrict__ gir = gi + (size_t)(b * TT + t) * H3;
            const float* __restrict__ ghr = gh + b * H3;
            const float r  = sigmoidf_(gir[j]      + ghr[j]      + bhs[j]);
            const float z  = sigmoidf_(gir[HH + j] + ghr[HH + j] + bhs[HH + j]);
            const float nv = fast_tanh(gir[2 * HH + j] +
                                       r * (ghr[2 * HH + j] + bhs[2 * HH + j]));
            const float hn = (1.0f - z) * nv + z * hf[idx];
            hf[idx] = hn;
            hb[idx] = (__bf16)hn;
            out[(size_t)(b * TT + t) * HH + j] = hn;
        }
        __syncthreads();
    }

    for (int idx = tid; idx < BB * HH; idx += 768) hT[idx] = hf[idx];
}

// ---------------------------------------------------------------------------
// pos[b][j] = i s.t. targets[b][i]==j (targets is a permutation per batch).
// dec_src[b][t] = b*T + targets[b][(t-1) mod T]  (teacher-forcing gather rows)
// ---------------------------------------------------------------------------
__global__ __launch_bounds__(256) void setup_idx(
    const int* __restrict__ targets, int* __restrict__ pos,
    int* __restrict__ dec_src)
{
    const int b = blockIdx.x;
    const int i = threadIdx.x;
    const int tgt = targets[b * TT + i];
    pos[b * TT + tgt] = i;
    const int prev = targets[b * TT + ((i + TT - 1) & (TT - 1))];
    dec_src[b * TT + i] = b * TT + prev;
}

// ---------------------------------------------------------------------------
// logits[b][t][s] = mask ? sum_a v[a]*tanh(q[b,t,a] + k[b,s,a]) : -1e9
// Block = (b, 8 query rows); k stored transposed kT[b][a][s] so the a-loop is
// coalesced over s lanes, and each kT load feeds 8 tanh/fma ops.
// mask = (pos[b][s] >= t)  ==  "s not yet emitted before step t".
// ---------------------------------------------------------------------------
__global__ __launch_bounds__(256) void attn_scores(
    const float* __restrict__ q, const float* __restrict__ kT,
    const float* __restrict__ v, const int* __restrict__ pos,
    float* __restrict__ out)
{
    __shared__ float qs[TQ][HH];
    __shared__ float vs[HH];
    const int b  = blockIdx.x / (TT / TQ);
    const int t0 = (blockIdx.x % (TT / TQ)) * TQ;
    const int tid = threadIdx.x;

    for (int idx = tid; idx < TQ * HH; idx += 256) {
        const int u = idx >> 8;
        const int a = idx & 255;
        qs[u][a] = q[(size_t)(b * TT + t0 + u) * HH + a];
    }
    vs[tid] = v[tid];
    __syncthreads();

    const int s = tid;
    const float* __restrict__ kc = kT + (size_t)b * (size_t)(HH * TT) + s;
    float acc[TQ] = {0.f, 0.f, 0.f, 0.f, 0.f, 0.f, 0.f, 0.f};
    #pragma unroll 2
    for (int a = 0; a < HH; ++a) {
        const float kv = kc[(size_t)a * TT];
        const float va = vs[a];
        #pragma unroll
        for (int u = 0; u < TQ; ++u)
            acc[u] = fmaf(va, fast_tanh(qs[u][a] + kv), acc[u]);
    }

    const int ps = pos[b * TT + s];
    #pragma unroll
    for (int u = 0; u < TQ; ++u)
        out[(size_t)(b * TT + t0 + u) * TT + s] = (ps >= t0 + u) ? acc[u] : -1.0e9f;
}

// ---------------------------------------------------------------------------
extern "C" void kernel_launch(void* const* d_in, const int* in_sizes, int n_in,
                              void* d_out, int out_size, void* d_ws, size_t ws_size,
                              hipStream_t stream)
{
    (void)in_sizes; (void)n_in; (void)out_size; (void)ws_size;

    const float* inputs  = (const float*)d_in[0];   // [B,T,D]
    const int*   targets = (const int*)  d_in[1];   // [B,T]
    const float* W_enc   = (const float*)d_in[2];   // [H,D]
    const float* b_enc   = (const float*)d_in[3];   // [H]
    const float* enc_Wih = (const float*)d_in[4];   // [3H,H]
    const float* enc_Whh = (const float*)d_in[5];   // [3H,H]
    const float* enc_bih = (const float*)d_in[6];   // [3H]
    const float* enc_bhh = (const float*)d_in[7];   // [3H]
    const float* dec_Wih = (const float*)d_in[8];
    const float* dec_Whh = (const float*)d_in[9];
    const float* dec_bih = (const float*)d_in[10];
    const float* dec_bhh = (const float*)d_in[11];
    const float* Wq      = (const float*)d_in[12];  // [H,H]
    const float* Wk      = (const float*)d_in[13];  // [H,H]
    const float* vvec    = (const float*)d_in[14];  // [H]
    float* logits = (float*)d_out;                  // [B,T,T]

    // Workspace layout (~33.6 MB)
    char* ws = (char*)d_ws;
    size_t off = 0;
    auto alloc = [&](size_t bytes) {
        void* p = ws + off;
        off = (off + bytes + 255) & ~(size_t)255;
        return p;
    };
    float* enc_in  = (float*)alloc((size_t)BB * TT * HH * 4);  // 4 MB
    float* gibuf   = (float*)alloc((size_t)BB * TT * H3 * 4);  // 12 MB (reused enc/dec)
    float* enc_out = (float*)alloc((size_t)BB * TT * HH * 4);  // 4 MB
    float* dec_out = (float*)alloc((size_t)BB * TT * HH * 4);  // 4 MB
    float* qbuf    = (float*)alloc((size_t)BB * TT * HH * 4);  // 4 MB
    float* kTbuf   = (float*)alloc((size_t)BB * TT * HH * 4);  // 4 MB
    float* hTbuf   = (float*)alloc((size_t)BB * HH * 4);
    int*   pos     = (int*)  alloc((size_t)BB * TT * 4);
    int*   dec_src = (int*)  alloc((size_t)BB * TT * 4);

    const size_t gru_lds = (size_t)(BB * HH + BB * H3 + H3) * 4
                         + (size_t)BB * HH * 2;                // 76800 B

    // 0) permutation positions + decoder gather indices
    setup_idx<<<BB, TT, 0, stream>>>(targets, pos, dec_src);

    // 1) enc_in = inputs @ W_enc^T + b_enc          (M=4096,K=128,N=256)
    gemm_bf16_wmma<<<128, 256, 0, stream>>>(inputs, W_enc, b_enc, enc_in,
                                            BB * TT, HH, DD, nullptr, 0);
    // 2) gi = enc_in @ enc_Wih^T + enc_bih          (M=4096,K=256,N=768)
    gemm_bf16_wmma<<<384, 256, 0, stream>>>(enc_in, enc_Wih, enc_bih, gibuf,
                                            BB * TT, H3, HH, nullptr, 0);
    // 3) encoder GRU scan -> enc_out, hT
    gru_scan<<<1, 768, gru_lds, stream>>>(gibuf, enc_Whh, enc_bhh, nullptr,
                                          enc_out, hTbuf);
    // 4) gi = enc_in[gathered rows] @ dec_Wih^T + dec_bih
    gemm_bf16_wmma<<<384, 256, 0, stream>>>(enc_in, dec_Wih, dec_bih, gibuf,
                                            BB * TT, H3, HH, dec_src, 0);
    // 5) decoder GRU scan (h0 = hT) -> dec_out
    gru_scan<<<1, 768, gru_lds, stream>>>(gibuf, dec_Whh, dec_bhh, hTbuf,
                                          dec_out, hTbuf);
    // 6) q = dec_out @ Wq^T                          (M=4096,K=256,N=256)
    gemm_bf16_wmma<<<128, 256, 0, stream>>>(dec_out, Wq, nullptr, qbuf,
                                            BB * TT, HH, HH, nullptr, 0);
    // 7) kT[b][a][s] = (enc_out @ Wk^T) transposed
    gemm_bf16_wmma<<<128, 256, 0, stream>>>(enc_out, Wk, nullptr, kTbuf,
                                            BB * TT, HH, HH, nullptr, 1);
    // 8) masked additive-attention scores (8 query rows per block)
    attn_scores<<<BB * (TT / TQ), 256, 0, stream>>>(qbuf, kTbuf, vvec, pos, logits);
}
